// SEGNNModel_15264313770286
// MI455X (gfx1250) — matrix operand order repaired and checked by reading, not compiled, not run on hardware
//
#include <hip/hip_runtime.h>
#include <hip/hip_bf16.h>

typedef __attribute__((ext_vector_type(16))) _Float16 v16h;
typedef __attribute__((ext_vector_type(8)))  float    v8f;

#define CG  64
#define CEPS 1e-5f
#define MSG_SZ 10092
#define UPD_SZ 10028
#define PACK_TILES 32           // tiles per weight set
#define PACK_SET   (PACK_TILES * 512)   // halves per weight set
#define NSETS 6                 // 3 msg + 3 upd

__device__ __forceinline__ float sig_(float x) { return 1.f / (1.f + __expf(-x)); }

// ---------------- WMMA fragment helpers (CDNA5 16x16x32 f16 layouts) ----------------
// A (16xK tile): lane<16 -> M=lane, halves 0..7 = K(kh+0..7), halves 8..15 = K(kh+16..23), kh=8*(lane>=16).
// K is a compile-time constant from {16,32,48,96}; all bound checks fold per-half.
template <int K>
__device__ __forceinline__ v16h a_frag(const float* __restrict__ src, int ld, int kbase) {
  const int lane = threadIdx.x & 31;
  const int m = lane & 15;
  const int kh = (lane >> 4) << 3;  // 0 or 8
  v16h a;
#pragma unroll
  for (int h = 0; h < 16; ++h) {
    const int kmin = kbase + ((h < 8) ? h : (8 + h));  // k when kh==0
    float f;
    if (kmin + 8 < K) {          // valid even for kh==8  -> unconditional load
      f = src[m * ld + kmin + kh];
    } else if (kmin >= K) {      // out of range even for kh==0 -> constant zero
      f = 0.0f;
    } else {                     // (never taken for K in {16,32,48,96} with 32-aligned kbase)
      int k = kmin + kh;
      f = (k < K) ? src[m * ld + k] : 0.0f;
    }
    a[h] = (_Float16)f;
  }
  return a;
}

// packed-B gemm: B tiles prepacked as [tile][lane][16 halves]; one v16h load per fragment.
template <int K>
__device__ __forceinline__ v8f gemmP(const float* __restrict__ feat, int ld,
                                     const _Float16* __restrict__ pw, int tbase, int ktiles,
                                     int nt, v8f acc) {
  const int lane = threadIdx.x & 31;
  constexpr int KT = (K + 31) / 32;
#pragma unroll
  for (int kt = 0; kt < KT; ++kt) {
    v16h a = a_frag<K>(feat, ld, kt * 32);
    v16h b = *(const v16h*)(pw + (size_t)(tbase + nt * ktiles + kt) * 512 + lane * 16);
    acc = __builtin_amdgcn_wmma_f32_16x16x32_f16(false, a, false, b, (short)0, acc, false, false);
  }
  return acc;
}

__device__ __forceinline__ void w_offsets(int KS, int* o) {
  o[0] = 0;
  o[1] = o[0] + KS * 42;   // Ws0
  o[2] = o[1] + 28 * 42;   // Ws1
  o[3] = o[2] + 16 * 42;   // Ws2
  o[4] = o[3] + 28 * 14;   // Wv0
  o[5] = o[4] + KS * 14;   // Wv1
  o[6] = o[5] + 28 * 14;   // Wv2
  o[7] = o[6] + 16 * 14;   // Wv3
  o[8] = o[7] + 28 * 14;   // Wv4
  o[9] = o[8] + 16 * 8;    // Wt0
  o[10] = o[9] + KS * 8;   // Wt1
  o[11] = o[10] + 28 * 8;  // Wt2
  o[12] = o[11] + 16 * 8;  // Wt3 -> Wg
}

// ---------------- weight repack: f32 row-major -> f16 B-fragment tiles ----------------
// Packed tile map (tile index within a set): PO[w] + nt*ktiles + kt
// w: 0:Ws0(K96,N3t) 1:Ws1 2:Ws2 3:Wv0 4:Wv1 5:Wv2 6:Wv3 7:Wv4 8:Wt0 9:Wt1 10:Wt2 11:Wt3 12:Wg
__global__ void repack_kernel(const float* __restrict__ w_msg, const float* __restrict__ w_upd,
                              _Float16* __restrict__ packed) {
  int idx = blockIdx.x * blockDim.x + threadIdx.x;
  if (idx >= NSETS * PACK_SET) return;
  int set = idx / PACK_SET;
  int rem = idx % PACK_SET;
  int t = rem / 512;
  int e = rem % 512;
  int lane = e / 16;
  int h = e % 16;
  bool isMsg = set < 3;
  int KS = isMsg ? 85 : 84;
  const float* W = isMsg ? (w_msg + (size_t)set * MSG_SZ) : (w_upd + (size_t)(set - 3) * UPD_SZ);
  const int PO[14] = {0, 9, 12, 15, 16, 19, 20, 21, 22, 23, 26, 27, 28, 32};
  const int KTn[13] = {3, 1, 1, 1, 3, 1, 1, 1, 1, 3, 1, 1, 2};
  const int Nd[13] = {42, 42, 42, 14, 14, 14, 14, 14, 8, 8, 8, 8, 22};
  int Kd[13] = {KS, 28, 16, 28, KS, 28, 16, 28, 16, KS, 28, 16, 42};
  int O[13];
  w_offsets(KS, O);
  int w = 12;
  for (int i = 0; i < 13; ++i)
    if (t < PO[i + 1]) { w = i; break; }
  int local = t - PO[w];
  int kt = local % KTn[w];
  int nt = local / KTn[w];
  int n = nt * 16 + (lane & 15);
  int k = kt * 32 + ((lane >> 4) << 3) + ((h < 8) ? h : (8 + h));
  float v = (k < Kd[w] && n < Nd[w]) ? W[O[w] + k * Nd[w] + n] : 0.0f;
  packed[idx] = (_Float16)v;
}

// ---------------- shared tensor-product + gate (per 16-row tile, one wave) ----------------
// Outputs: sS = silu(s_out) [16][48] (cols 42..47 zero), oV gated vectors [16][14][3], oT gated [16][8][9]
__device__ __forceinline__ void tp_and_gate(const _Float16* __restrict__ pw,
                                            const float* sSS, const float* sVV, const float* sTT,
                                            float* sPV, float* sPT, float* sTA1,
                                            const float* sA0, const float* sA1, const float* sA2,
                                            float* sS, float* sG, float* oV, float* oT,
                                            float* tGb, float* bK) {
  const int lane = threadIdx.x & 31;
  __syncthreads();
  // ---- projected inputs: PV[c]=v.a1, PT[c]=t:a2, TA1_i[c]=sum_j t[c,i,j] a1[j]
  for (int idx = lane; idx < 16 * 32; idx += 32) {
    int m = idx >> 5;
    float acc = 0.f;
#pragma unroll
    for (int i = 0; i < 3; ++i) acc += sVV[i * 512 + idx] * sA1[m * 3 + i];  // pad cols are 0
    sPV[idx] = acc;
  }
  for (int idx = lane; idx < 256; idx += 32) {
    int m = idx >> 4;
    float acc = 0.f;
#pragma unroll
    for (int ij = 0; ij < 9; ++ij) acc += sTT[ij * 256 + idx] * sA2[m * 9 + ij];
    sPT[idx] = acc;
#pragma unroll
    for (int i = 0; i < 3; ++i) {
      float a = 0.f;
#pragma unroll
      for (int j = 0; j < 3; ++j) a += sTT[(i * 3 + j) * 256 + idx] * sA1[m * 3 + j];
      sTA1[i * 256 + idx] = a;
    }
  }
  __syncthreads();

  const v8f ZV = {0.f, 0.f, 0.f, 0.f, 0.f, 0.f, 0.f, 0.f};
  const int n16 = lane & 15;
  const int mh = (lane >> 4) << 3;

  // ---- scalar channel: s_out = (ss@Ws0)*a0 + pv@Ws1 + pt@Ws2
#pragma unroll
  for (int nt = 0; nt < 3; ++nt) {
    v8f c0 = gemmP<96>(sSS, 96, pw, 0, 3, nt, ZV);
    v8f c1 = gemmP<32>(sPV, 32, pw, 9, 1, nt, ZV);
    c1 = gemmP<16>(sPT, 16, pw, 12, 1, nt, c1);
    int n = nt * 16 + n16;
    if (n < 42) {
#pragma unroll
      for (int r = 0; r < 8; ++r) { int m = r + mh; sS[m * 48 + n] = c0[r] * sA0[m] + c1[r]; }
    }
  }
  // zero pad cols 42..47 of the gate input
  for (int idx = lane; idx < 16 * 6; idx += 32) sS[(idx / 6) * 48 + 42 + (idx % 6)] = 0.f;
  __syncthreads();
  // ---- gate g = sigmoid(s_out @ Wg)  (22 outs: 14 vec + 8 tens)
#pragma unroll
  for (int nt = 0; nt < 2; ++nt) {
    v8f c = gemmP<48>(sS, 48, pw, 28, 2, nt, ZV);
    int n = nt * 16 + n16;
    if (n < 22) {
#pragma unroll
      for (int r = 0; r < 8; ++r) { int m = r + mh; sG[m * 22 + n] = sig_(c[r]); }
    }
  }
  __syncthreads();
  // silu(s_out) in place (silu(0)=0 so pad cols stay zero)
  for (int idx = lane; idx < 16 * 48; idx += 32) { float x = sS[idx]; sS[idx] = x * sig_(x); }

  // ---- vector channel
  v8f vB = gemmP<96>(sSS, 96, pw, 16, 3, 0, ZV);
  v8f vA[3], vC[3], vD[3], vE[3];
#pragma unroll
  for (int i = 0; i < 3; ++i) {
    vA[i] = gemmP<32>(sVV + i * 512, 32, pw, 15, 1, 0, ZV);
    vC[i] = gemmP<32>(sVV + i * 512, 32, pw, 19, 1, 0, ZV);
    vD[i] = gemmP<16>(sTA1 + i * 256, 16, pw, 20, 1, 0, ZV);
    vE[i] = gemmP<32>(sVV + i * 512, 32, pw, 21, 1, 0, ZV);
  }
  if (n16 < 14) {
#pragma unroll
    for (int r = 0; r < 8; ++r) {
      int m = r + mh;
      float ax = sA1[m * 3], ay = sA1[m * 3 + 1], az = sA1[m * 3 + 2];
      float Cx = vC[0][r], Cy = vC[1][r], Cz = vC[2][r];
      float cr[3] = {Cy * az - Cz * ay, Cz * ax - Cx * az, Cx * ay - Cy * ax};
      float g = sG[m * 22 + n16];
#pragma unroll
      for (int i = 0; i < 3; ++i) {
        float esum = vE[0][r] * sA2[m * 9 + i] + vE[1][r] * sA2[m * 9 + 3 + i] + vE[2][r] * sA2[m * 9 + 6 + i];
        float val = vA[i][r] * sA0[m] + vB[r] * sA1[m * 3 + i] + cr[i] + vD[i][r] + esum;
        oV[(m * 14 + n16) * 3 + i] = val * g;
      }
    }
  }

  // ---- tensor channel
  {
    v8f cG = gemmP<96>(sSS, 96, pw, 23, 3, 0, ZV);
    if (n16 < 8) {
#pragma unroll
      for (int r = 0; r < 8; ++r) { int m = r + mh; tGb[m * 8 + n16] = cG[r]; }
    }
  }
#pragma unroll
  for (int ij = 0; ij < 9; ++ij) {
    v8f cF = gemmP<16>(sTT + ij * 256, 16, pw, 22, 1, 0, ZV);
    v8f cK = gemmP<16>(sTT + ij * 256, 16, pw, 27, 1, 0, ZV);
    if (n16 < 8) {
#pragma unroll
      for (int r = 0; r < 8; ++r) {
        int m = r + mh;
        oT[(m * 8 + n16) * 9 + ij] = cF[r] * sA0[m] + tGb[m * 8 + n16] * sA2[m * 9 + ij];
        bK[(m * 8 + n16) * 9 + ij] = cK[r];
      }
    }
  }
  v8f vH[3];
#pragma unroll
  for (int i = 0; i < 3; ++i) vH[i] = gemmP<32>(sVV + i * 512, 32, pw, 26, 1, 0, ZV);
  if (n16 < 8) {
#pragma unroll
    for (int r = 0; r < 8; ++r) {
      int m = r + mh;
      const float* a2m = &sA2[m * 9];
      float a1m[3] = {sA1[m * 3], sA1[m * 3 + 1], sA1[m * 3 + 2]};
      float H[3] = {vH[0][r], vH[1][r], vH[2][r]};
      float trH = H[0] * a1m[0] + H[1] * a1m[1] + H[2] * a1m[2];
      const float* Kp = &bK[(m * 8 + n16) * 9];
      float M[3][3];
      float trM = 0.f;
#pragma unroll
      for (int i2 = 0; i2 < 3; ++i2) {
#pragma unroll
        for (int k = 0; k < 3; ++k)
          M[i2][k] = Kp[i2 * 3] * a2m[k] + Kp[i2 * 3 + 1] * a2m[3 + k] + Kp[i2 * 3 + 2] * a2m[6 + k];
        trM += M[i2][i2];
      }
      float g = sG[m * 22 + 14 + n16];
      float sub = (trH + trM) * (1.f / 3.f);
#pragma unroll
      for (int i2 = 0; i2 < 3; ++i2) {
#pragma unroll
        for (int k = 0; k < 3; ++k) {
          float val = oT[(m * 8 + n16) * 9 + i2 * 3 + k]
                    + 0.5f * (H[i2] * a1m[k] + H[k] * a1m[i2])
                    + 0.5f * (M[i2][k] + M[k][i2]);
          if (i2 == k) val -= sub;
          oT[(m * 8 + n16) * 9 + i2 * 3 + k] = val * g;
        }
      }
    }
  }
  __syncthreads();
}

// ---------------- small kernels ----------------
__global__ void cnt_kernel(const int* __restrict__ batch, float* __restrict__ cnt, int Nn) {
  int n = blockIdx.x * blockDim.x + threadIdx.x;
  if (n < Nn) atomicAdd(&cnt[batch[n]], 1.f);
}

__global__ void edge_attrs_kernel(const float* __restrict__ pos, const int* __restrict__ row,
                                  const int* __restrict__ col, float* __restrict__ a1e,
                                  float* __restrict__ a2e, float* __restrict__ na1,
                                  float* __restrict__ na2, float* __restrict__ deg, int E) {
  int e = blockIdx.x * blockDim.x + threadIdx.x;
  if (e >= E) return;
  int r = row[e], c = col[e];
  float dx = pos[r * 3] - pos[c * 3];
  float dy = pos[r * 3 + 1] - pos[c * 3 + 1];
  float dz = pos[r * 3 + 2] - pos[c * 3 + 2];
  float nrm = fmaxf(sqrtf(dx * dx + dy * dy + dz * dz), 1e-6f);
  float u[3] = {dx / nrm, dy / nrm, dz / nrm};
  const float s3 = 1.7320508075688772f, s75 = 2.7386127875258306f;
#pragma unroll
  for (int i = 0; i < 3; ++i) {
    float a = s3 * u[i];
    a1e[e * 3 + i] = a;
    atomicAdd(&na1[c * 3 + i], a);
  }
#pragma unroll
  for (int i = 0; i < 3; ++i)
#pragma unroll
    for (int j = 0; j < 3; ++j) {
      float a = s75 * (u[i] * u[j] - ((i == j) ? (1.f / 3.f) : 0.f));
      a2e[e * 9 + i * 3 + j] = a;
      atomicAdd(&na2[c * 9 + i * 3 + j], a);
    }
  atomicAdd(&deg[c], 1.f);
}

__global__ void node_attr_norm_kernel(float* __restrict__ na1, float* __restrict__ na2,
                                      const float* __restrict__ deg, float* __restrict__ na0, int Nn) {
  int n = blockIdx.x * blockDim.x + threadIdx.x;
  if (n >= Nn) return;
  float d = fmaxf(deg[n], 1.f);
  float inv = 1.f / d;
#pragma unroll
  for (int i = 0; i < 3; ++i) na1[n * 3 + i] *= inv;
#pragma unroll
  for (int j = 0; j < 9; ++j) na2[n * 9 + j] *= inv;
  na0[n] = (deg[n] > 0.f) ? 1.f : 0.f;
}

__global__ void embed_kernel(const int* __restrict__ x, const float* __restrict__ emb,
                             const float* __restrict__ w_emb, const float* __restrict__ na0,
                             const float* __restrict__ na1, const float* __restrict__ na2,
                             float* __restrict__ hs, float* __restrict__ hv, float* __restrict__ ht,
                             int Nn) {
  int n = blockIdx.x * blockDim.x + threadIdx.x;
  if (n >= Nn) return;
  const float* e = emb + x[n] * 32;
  const float* Ws = w_emb;
  const float* Wv = w_emb + 32 * 42;
  const float* Wt = Wv + 32 * 14;
  float a0 = na0[n];
  for (int c = 0; c < 42; ++c) {
    float s = 0.f;
    for (int k = 0; k < 32; ++k) s += e[k] * Ws[k * 42 + c];
    hs[n * 42 + c] = s * a0;
  }
  for (int c = 0; c < 14; ++c) {
    float s = 0.f;
    for (int k = 0; k < 32; ++k) s += e[k] * Wv[k * 14 + c];
#pragma unroll
    for (int i = 0; i < 3; ++i) hv[(n * 14 + c) * 3 + i] = s * na1[n * 3 + i];
  }
  for (int c = 0; c < 8; ++c) {
    float s = 0.f;
    for (int k = 0; k < 32; ++k) s += e[k] * Wt[k * 8 + c];
#pragma unroll
    for (int j = 0; j < 9; ++j) ht[(n * 8 + c) * 9 + j] = s * na2[n * 9 + j];
  }
}

// ---------------- message kernel: one wave per 16-edge tile ----------------
__global__ __launch_bounds__(32) void msg_kernel(
    const float* __restrict__ hs, const float* __restrict__ hv, const float* __restrict__ ht,
    const int* __restrict__ row, const int* __restrict__ col, const float* __restrict__ ew,
    const float* __restrict__ a1e, const float* __restrict__ a2e,
    const _Float16* __restrict__ pw,
    float* __restrict__ agg_s, float* __restrict__ agg_v, float* __restrict__ agg_t, int E) {
  __shared__ float sSS[16 * 96], sVV[3 * 16 * 32], sTT[9 * 256];
  __shared__ float sPV[16 * 32], sPT[256], sTA1[3 * 256];
  __shared__ float sA0[16], sA1[16 * 3], sA2[16 * 9];
  __shared__ float sS[16 * 48], sG[16 * 22], oV[16 * 42], oT[16 * 72];
  __shared__ float tGb[16 * 8], bK[16 * 72];
  __shared__ int sRow[16], sCol[16];
  const int lane = threadIdx.x;
  const int e0 = blockIdx.x * 16;
  if (lane < 16) {
    int e = e0 + lane;
    if (e >= E) e = E - 1;
    sRow[lane] = row[e];
    sCol[lane] = col[e];
    sA0[lane] = 1.0f;  // a0e == 1 for all edges
#pragma unroll
    for (int i = 0; i < 3; ++i) sA1[lane * 3 + i] = a1e[e * 3 + i];
#pragma unroll
    for (int j = 0; j < 9; ++j) sA2[lane * 9 + j] = a2e[e * 9 + j];
  }
  __syncthreads();
  for (int idx = lane; idx < 16 * 96; idx += 32) {
    int m = idx / 96, k = idx % 96;
    float v = 0.f;
    if (k < 42) v = hs[sRow[m] * 42 + k];
    else if (k < 84) v = hs[sCol[m] * 42 + (k - 42)];
    else if (k == 84) { int e = e0 + m; v = (e < E) ? ew[e] : 0.f; }
    sSS[idx] = v;
  }
  for (int idx = lane; idx < 3 * 512; idx += 32) {
    int i = idx / 512, r2 = idx & 511, m = r2 >> 5, c = r2 & 31;
    float v = 0.f;
    if (c < 14) v = hv[(sRow[m] * 14 + c) * 3 + i];
    else if (c < 28) v = hv[(sCol[m] * 14 + (c - 14)) * 3 + i];
    sVV[idx] = v;
  }
  for (int idx = lane; idx < 9 * 256; idx += 32) {
    int ij = idx >> 8, r2 = idx & 255, m = r2 >> 4, c = r2 & 15;
    sTT[idx] = (c < 8) ? ht[(sRow[m] * 8 + c) * 9 + ij] : ht[(sCol[m] * 8 + (c - 8)) * 9 + ij];
  }
  tp_and_gate(pw, sSS, sVV, sTT, sPV, sPT, sTA1, sA0, sA1, sA2, sS, sG, oV, oT, tGb, bK);
  for (int idx = lane; idx < 16 * 42; idx += 32) {
    int m = idx / 42, c = idx % 42;
    if (e0 + m < E) atomicAdd(&agg_s[sCol[m] * 42 + c], sS[m * 48 + c]);
  }
  for (int idx = lane; idx < 16 * 42; idx += 32) {
    int m = idx / 42;
    if (e0 + m < E) atomicAdd(&agg_v[sCol[m] * 42 + idx % 42], oV[idx]);
  }
  for (int idx = lane; idx < 16 * 72; idx += 32) {
    int m = idx / 72;
    if (e0 + m < E) atomicAdd(&agg_t[sCol[m] * 72 + idx % 72], oT[idx]);
  }
}

// ---------------- update kernel: one wave per 16-node tile ----------------
__global__ __launch_bounds__(32) void upd_kernel(
    float* __restrict__ hs, float* __restrict__ hv, float* __restrict__ ht,
    const float* __restrict__ agg_s, const float* __restrict__ agg_v, const float* __restrict__ agg_t,
    const float* __restrict__ na0, const float* __restrict__ na1, const float* __restrict__ na2,
    const int* __restrict__ batch, const _Float16* __restrict__ pw,
    float* __restrict__ st_s, float* __restrict__ st_s2, float* __restrict__ st_v,
    float* __restrict__ st_t, int Nn) {
  __shared__ float sSS[16 * 96], sVV[3 * 16 * 32], sTT[9 * 256];
  __shared__ float sPV[16 * 32], sPT[256], sTA1[3 * 256];
  __shared__ float sA0[16], sA1[16 * 3], sA2[16 * 9];
  __shared__ float sS[16 * 48], sG[16 * 22], oV[16 * 42], oT[16 * 72];
  __shared__ float tGb[16 * 8], bK[16 * 72];
  const int lane = threadIdx.x;
  const int n0 = blockIdx.x * 16;
  if (lane < 16) {
    int n = n0 + lane;
    if (n >= Nn) n = Nn - 1;
    sA0[lane] = na0[n];
#pragma unroll
    for (int i = 0; i < 3; ++i) sA1[lane * 3 + i] = na1[n * 3 + i];
#pragma unroll
    for (int j = 0; j < 9; ++j) sA2[lane * 9 + j] = na2[n * 9 + j];
  }
  for (int idx = lane; idx < 16 * 96; idx += 32) {
    int m = idx / 96, k = idx % 96;
    int n = n0 + m;
    if (n >= Nn) n = Nn - 1;
    float v = 0.f;
    if (k < 42) v = hs[n * 42 + k];
    else if (k < 84) v = agg_s[n * 42 + (k - 42)];
    sSS[idx] = v;
  }
  for (int idx = lane; idx < 3 * 512; idx += 32) {
    int i = idx / 512, r2 = idx & 511, m = r2 >> 5, c = r2 & 31;
    int n = n0 + m;
    if (n >= Nn) n = Nn - 1;
    float v = 0.f;
    if (c < 14) v = hv[(n * 14 + c) * 3 + i];
    else if (c < 28) v = agg_v[(n * 14 + (c - 14)) * 3 + i];
    sVV[idx] = v;
  }
  for (int idx = lane; idx < 9 * 256; idx += 32) {
    int ij = idx >> 8, r2 = idx & 255, m = r2 >> 4, c = r2 & 15;
    int n = n0 + m;
    if (n >= Nn) n = Nn - 1;
    sTT[idx] = (c < 8) ? ht[(n * 8 + c) * 9 + ij] : agg_t[n * 72 + (c - 8) * 9 + ij];
  }
  tp_and_gate(pw, sSS, sVV, sTT, sPV, sPT, sTA1, sA0, sA1, sA2, sS, sG, oV, oT, tGb, bK);
  // residual + instance-norm statistics
  for (int idx = lane; idx < 16 * 42; idx += 32) {
    int m = idx / 42, c = idx % 42, n = n0 + m;
    if (n < Nn) {
      float v = hs[n * 42 + c] + sS[m * 48 + c];
      hs[n * 42 + c] = v;
      int g = batch[n];
      atomicAdd(&st_s[g * 42 + c], v);
      atomicAdd(&st_s2[g * 42 + c], v * v);
    }
  }
  for (int idx = lane; idx < 16 * 14; idx += 32) {
    int m = idx / 14, c = idx % 14, n = n0 + m;
    if (n < Nn) {
      float sq = 0.f;
#pragma unroll
      for (int i = 0; i < 3; ++i) {
        float v = hv[(n * 14 + c) * 3 + i] + oV[(m * 14 + c) * 3 + i];
        hv[(n * 14 + c) * 3 + i] = v;
        sq += v * v;
      }
      atomicAdd(&st_v[batch[n] * 14 + c], sq);
    }
  }
  for (int idx = lane; idx < 16 * 8; idx += 32) {
    int m = idx / 8, c = idx % 8, n = n0 + m;
    if (n < Nn) {
      float sq = 0.f;
#pragma unroll
      for (int j = 0; j < 9; ++j) {
        float v = ht[(n * 8 + c) * 9 + j] + oT[(m * 8 + c) * 9 + j];
        ht[(n * 8 + c) * 9 + j] = v;
        sq += v * v;
      }
      atomicAdd(&st_t[batch[n] * 8 + c], sq);
    }
  }
}

__global__ void inorm_kernel(float* __restrict__ hs, float* __restrict__ hv, float* __restrict__ ht,
                             const float* __restrict__ st_s, const float* __restrict__ st_s2,
                             const float* __restrict__ st_v, const float* __restrict__ st_t,
                             const float* __restrict__ cnt, const int* __restrict__ batch, int Nn) {
  int n = blockIdx.x * blockDim.x + threadIdx.x;
  if (n >= Nn) return;
  int g = batch[n];
  float inv = 1.f / fmaxf(cnt[g], 1.f);
  for (int c = 0; c < 42; ++c) {
    float mean = st_s[g * 42 + c] * inv;
    float var = fmaxf(st_s2[g * 42 + c] * inv - mean * mean, 0.f);
    hs[n * 42 + c] = (hs[n * 42 + c] - mean) * rsqrtf(var + CEPS);
  }
  for (int c = 0; c < 14; ++c) {
    float s = rsqrtf(st_v[g * 14 + c] * inv + CEPS);
#pragma unroll
    for (int i = 0; i < 3; ++i) hv[(n * 14 + c) * 3 + i] *= s;
  }
  for (int c = 0; c < 8; ++c) {
    float s = rsqrtf(st_t[g * 8 + c] * inv + CEPS);
#pragma unroll
    for (int j = 0; j < 9; ++j) ht[(n * 8 + c) * 9 + j] *= s;
  }
}

__global__ void pool_kernel(const float* __restrict__ hs, const int* __restrict__ batch,
                            float* __restrict__ pooled, int Nn) {
  int idx = blockIdx.x * blockDim.x + threadIdx.x;
  if (idx >= Nn * 42) return;
  int n = idx / 42, c = idx % 42;
  atomicAdd(&pooled[batch[n] * 42 + c], hs[idx]);
}

__global__ void head_kernel(const float* __restrict__ pooled, const float* __restrict__ cnt,
                            const float* __restrict__ W1, const float* __restrict__ W2,
                            float* __restrict__ out) {
  int g = threadIdx.x;
  if (g >= CG) return;
  float inv = 1.f / fmaxf(cnt[g], 1.f);
  float p[42], t[42];
  for (int c = 0; c < 42; ++c) p[c] = pooled[g * 42 + c] * inv;
  for (int c = 0; c < 42; ++c) {
    float s = 0.f;
    for (int k = 0; k < 42; ++k) s += p[k] * W1[k * 42 + c];
    t[c] = s * sig_(s);
  }
  for (int o = 0; o < 16; ++o) {
    float s = 0.f;
    for (int c = 0; c < 42; ++c) s += t[c] * W2[c * 16 + o];
    out[g * 16 + o] = s;
  }
}

// ---------------- host launcher ----------------
extern "C" void kernel_launch(void* const* d_in, const int* in_sizes, int n_in,
                              void* d_out, int out_size, void* d_ws, size_t ws_size,
                              hipStream_t stream) {
  (void)n_in; (void)out_size; (void)ws_size;
  const int*   x      = (const int*)d_in[0];
  const float* pos    = (const float*)d_in[1];
  const int*   batch  = (const int*)d_in[2];
  const int*   eidx   = (const int*)d_in[3];
  const float* ew     = (const float*)d_in[4];
  const float* emb    = (const float*)d_in[5];
  const float* w_emb  = (const float*)d_in[6];
  const float* w_msg  = (const float*)d_in[7];
  const float* w_upd  = (const float*)d_in[8];
  const float* w_h1   = (const float*)d_in[9];
  const float* w_h2   = (const float*)d_in[10];
  float* out = (float*)d_out;

  const int Nn = in_sizes[0];
  const int E = in_sizes[4];
  const int* row = eidx;
  const int* col = eidx + E;

  float* ws = (float*)d_ws;
  size_t o = 0;
  float* na1 = ws + o; o += (size_t)Nn * 3;
  float* na2 = ws + o; o += (size_t)Nn * 9;
  float* deg = ws + o; o += (size_t)Nn;
  float* na0 = ws + o; o += (size_t)Nn;
  float* cnt = ws + o; o += CG;
  size_t zero1 = o;                       // [0, zero1) zeroed at start
  float* a1e = ws + o; o += (size_t)E * 3;
  float* a2e = ws + o; o += (size_t)E * 9;
  float* hs = ws + o;  o += (size_t)Nn * 42;
  float* hv = ws + o;  o += (size_t)Nn * 42;
  float* ht = ws + o;  o += (size_t)Nn * 72;
  size_t zoff = o;                        // per-layer zero region
  float* agg_s = ws + o; o += (size_t)Nn * 42;
  float* agg_v = ws + o; o += (size_t)Nn * 42;
  float* agg_t = ws + o; o += (size_t)Nn * 72;
  float* st_s = ws + o;  o += (size_t)CG * 42;
  float* st_s2 = ws + o; o += (size_t)CG * 42;
  float* st_v = ws + o;  o += (size_t)CG * 14;
  float* st_t = ws + o;  o += (size_t)CG * 8;
  size_t zlen = o - zoff;
  float* pooled = ws + o; o += (size_t)CG * 42;
  o = (o + 7) & ~(size_t)7;               // 32B-align packed weights
  _Float16* packed = (_Float16*)(ws + o); o += (NSETS * PACK_SET) / 2;

  hipMemsetAsync(ws, 0, zero1 * sizeof(float), stream);
  repack_kernel<<<(NSETS * PACK_SET + 255) / 256, 256, 0, stream>>>(w_msg, w_upd, packed);
  cnt_kernel<<<(Nn + 255) / 256, 256, 0, stream>>>(batch, cnt, Nn);
  edge_attrs_kernel<<<(E + 255) / 256, 256, 0, stream>>>(pos, row, col, a1e, a2e, na1, na2, deg, E);
  node_attr_norm_kernel<<<(Nn + 255) / 256, 256, 0, stream>>>(na1, na2, deg, na0, Nn);
  embed_kernel<<<(Nn + 255) / 256, 256, 0, stream>>>(x, emb, w_emb, na0, na1, na2, hs, hv, ht, Nn);

  for (int l = 0; l < 3; ++l) {
    hipMemsetAsync(ws + zoff, 0, zlen * sizeof(float), stream);
    msg_kernel<<<(E + 15) / 16, 32, 0, stream>>>(hs, hv, ht, row, col, ew, a1e, a2e,
                                                 packed + (size_t)l * PACK_SET,
                                                 agg_s, agg_v, agg_t, E);
    upd_kernel<<<(Nn + 15) / 16, 32, 0, stream>>>(hs, hv, ht, agg_s, agg_v, agg_t,
                                                  na0, na1, na2, batch,
                                                  packed + (size_t)(3 + l) * PACK_SET,
                                                  st_s, st_s2, st_v, st_t, Nn);
    inorm_kernel<<<(Nn + 255) / 256, 256, 0, stream>>>(hs, hv, ht, st_s, st_s2, st_v, st_t,
                                                       cnt, batch, Nn);
  }
  hipMemsetAsync(pooled, 0, (size_t)CG * 42 * sizeof(float), stream);
  pool_kernel<<<(Nn * 42 + 255) / 256, 256, 0, stream>>>(hs, batch, pooled, Nn);
  head_kernel<<<1, 64, 0, stream>>>(pooled, cnt, w_h1, w_h2, out);
}
